// biLSTM_16784732193037
// MI455X (gfx1250) — matrix operand order repaired and checked by reading, not compiled, not run on hardware
//
#include <hip/hip_runtime.h>
#include <hip/hip_bf16.h>
#include <stdint.h>

// ---------------------------------------------------------------------------
// Bidirectional 2-layer LSTM (B=2048, T=512, D=1, H=36) + FC on last step.
// MI455X / gfx1250, wave32, V_WMMA_F32_16X16X32_F16.
//
// Per-direction recurrent step is cast as z = A(16x K) x B(K x 192):
//   N = 192 = 12 tiles of 16: gates [i|f|g|o], each padded 36->48 (3 tiles)
//       so tiles (k, k+3, k+6, k+9) share the same lane<->hidden mapping and
//       the gate combine is purely element-wise on accumulators.
//   K (layer0) = [h_prev 0..47 | x @48 | pad] -> 64   (2 WMMA K-chunks)
//   K (layer1) = [h_prev 0..47 | hf0 48..83 | hr0 84..119 | pad] -> 128 (4)
// Weights pre-permuted into the exact 32x16 B-operand register layout; the
// inner loop is ds_load_b128 (A/B) + v_wmma chains; c-state lives in f32
// accumulator registers for the whole scan. Activations are branch-free
// V_TANH_F32 (or v_exp/v_rcp fallback) -- the scan is latency-bound, so the
// per-step critical path is what matters.
// Layer-1 reverse needs only its first step (out[:, -1]) -> Tloop = 1.
// ---------------------------------------------------------------------------

typedef __attribute__((ext_vector_type(16))) _Float16 v16h;
typedef __attribute__((ext_vector_type(8)))  _Float16 v8h;
typedef __attribute__((ext_vector_type(8)))  float    v8f;

#define H36   36
#define TSEQ  512
#define BTOT  2048
#define NBT   128          // BTOT / 16 batch tiles
#define NTILE 12           // 4 gates * 3 (48/16) N-tiles
#define WSLOT 24576        // halves per packed-weight slot (12*4*32*16)

// ---------------- branch-free activations (no libm, no exec branches) ------
__device__ __forceinline__ float fast_tanh(float x) {
#if __has_builtin(__builtin_amdgcn_tanhf)
  return __builtin_amdgcn_tanhf(x);            // v_tanh_f32 (gfx1250 TRANS)
#else
  // tanh(x) = 1 - 2/(exp2(2*log2e*x) + 1); v_exp saturates cleanly at +/-inf
  const float t = __builtin_amdgcn_exp2f(x * 2.8853900818f);
  return 1.0f - 2.0f * __builtin_amdgcn_rcpf(t + 1.0f);
#endif
}
__device__ __forceinline__ float fast_sig(float x) {
  return 0.5f * fast_tanh(0.5f * x) + 0.5f;    // one transcendental
}

// --------------------------- weight packing --------------------------------
__global__ void pack_weights(
    const float* __restrict__ Wih0f, const float* __restrict__ Whh0f,
    const float* __restrict__ bih0f, const float* __restrict__ bhh0f,
    const float* __restrict__ Wih0r, const float* __restrict__ Whh0r,
    const float* __restrict__ bih0r, const float* __restrict__ bhh0r,
    const float* __restrict__ Wih1f, const float* __restrict__ Whh1f,
    const float* __restrict__ bih1f, const float* __restrict__ bhh1f,
    const float* __restrict__ Wih1r, const float* __restrict__ Whh1r,
    const float* __restrict__ bih1r, const float* __restrict__ bhh1r,
    _Float16* __restrict__ wpack, float* __restrict__ biasp)
{
  const int ld = blockIdx.x;               // 0=L0f 1=L0r 2=L1f 3=L1r
  const float *Wih, *Whh, *bih, *bhh;
  if      (ld == 0) { Wih=Wih0f; Whh=Whh0f; bih=bih0f; bhh=bhh0f; }
  else if (ld == 1) { Wih=Wih0r; Whh=Whh0r; bih=bih0r; bhh=bhh0r; }
  else if (ld == 2) { Wih=Wih1f; Whh=Whh1f; bih=bih1f; bhh=bhh1f; }
  else              { Wih=Wih1r; Whh=Whh1r; bih=bih1r; bhh=bhh1r; }
  const int layer = ld >> 1;
  const int KC    = layer ? 4 : 2;
  _Float16* wp = wpack + (size_t)ld * WSLOT;

  // B-operand layout (32x16, 16-bit):  lane l, element h ->
  //   K = c*32 + (l>=16 ? 16 : 0) + h,   N = tile*16 + (l & 15)
  const int nent = NTILE * KC * 512;
  for (int idx = threadIdx.x; idx < nent; idx += blockDim.x) {
    const int h    = idx & 15;
    const int lane = (idx >> 4) & 31;
    const int c    = (idx >> 9) % KC;
    const int tile = idx / (512 * KC);
    const int k    = c * 32 + ((lane >> 4) << 4) + h;
    const int gate = tile / 3;
    const int j    = (tile % 3) * 16 + (lane & 15);   // hidden unit (padded)
    float v = 0.f;
    if (j < H36) {
      const int go = gate * H36 + j;                  // original row 0..143
      if (layer == 0) {
        if      (k < H36)   v = Whh[go * H36 + k];
        else if (k == 48)   v = Wih[go];              // D == 1
      } else {
        if      (k < H36)             v = Whh[go * H36 + k];
        else if (k >= 48 && k < 120)  v = Wih[go * 72 + (k - 48)];
      }
    }
    wp[idx] = (_Float16)v;
  }
  for (int n = threadIdx.x; n < 192; n += blockDim.x) {
    const int tile = n / 16, gate = tile / 3;
    const int j = (tile % 3) * 16 + (n & 15);
    float v = 0.f;
    if (j < H36) { const int go = gate * H36 + j; v = bih[go] + bhh[go]; }
    biasp[ld * 192 + n] = v;
  }
}

// --------------------------- recurrent scan --------------------------------
template<int LAYER>
__global__ __launch_bounds__(32)
void lstm_scan(const float*    __restrict__ x,       // layer0 input (B,T)
               const uint32_t* __restrict__ g_h0f,   // layer1 inputs (dwords)
               const uint32_t* __restrict__ g_h0r,
               const _Float16* __restrict__ wpack,   // packed B weights
               const float*    __restrict__ biasp,   // permuted bias (192)
               uint32_t*       __restrict__ h0out,   // layer0 per-step h (f16)
               float*          __restrict__ h1out,   // layer1 final h (B,36)
               int dirrev, int Tloop)
{
  constexpr int KC = LAYER ? 4 : 2;
  constexpr int AW = 128;                       // A-tile width (halves)
  __shared__ alignas(32) _Float16 Wlds[NTILE * KC * 32 * 16];
  __shared__ alignas(16) _Float16 Atile[16 * AW];

  const int lane = threadIdx.x;
  const int bt   = blockIdx.x;
  const int b0   = bt * 16;
  const int mrow = lane & 15;
  const int hi16 = lane >> 4;

  { // weights -> LDS once (25/49 KB)
    const uint32_t* s = (const uint32_t*)wpack;
    uint32_t* d = (uint32_t*)Wlds;
    for (int i = lane; i < NTILE * KC * 256; i += 32) d[i] = s[i];
  }
  { // zero A-tile: h_prev = 0 and all padding columns
    uint32_t* d = (uint32_t*)Atile;
    for (int i = lane; i < 16 * AW / 2; i += 32) d[i] = 0u;
  }
  float bias[NTILE];
  #pragma unroll
  for (int tl = 0; tl < NTILE; ++tl) bias[tl] = biasp[tl * 16 + mrow];

  v8f cst[3];                                   // c-state, D-layout, f32
  #pragma unroll
  for (int k = 0; k < 3; ++k)
    #pragma unroll
    for (int e = 0; e < 8; ++e) cst[k][e] = 0.f;

  __syncthreads();

  for (int t = 0; t < Tloop; ++t) {
    const int tt = dirrev ? (TSEQ - 1 - t) : t;

    // -------- stage step inputs into A-tile (cols >= 48) --------
    if (LAYER == 0) {
      if (lane < 16) {
        const float xv = x[(size_t)(b0 + lane) * TSEQ + tt];
        Atile[lane * AW + 48] = (_Float16)xv;
      }
    } else {
      uint32_t* ad = (uint32_t*)Atile;
      const size_t base = ((size_t)tt * NBT + bt) * 384;
      for (int i = lane; i < 288; i += 32) {            // hf0 -> cols 48..83
        const int r = i / 18, d = i % 18;
        ad[r * 64 + 24 + d] = g_h0f[base + r * 24 + d];
      }
      for (int i = lane; i < 288; i += 32) {            // hr0 -> cols 84..119
        const int r = i / 18, d = i % 18;
        ad[r * 64 + 42 + d] = g_h0r[base + r * 24 + d];
      }
      if (t + 1 < Tloop) {                              // prefetch next step
        const size_t nb = ((size_t)(dirrev ? (TSEQ-2-t) : (t+1)) * NBT + bt) * 384;
        __builtin_prefetch((const char*)(g_h0f + nb) + lane * 64, 0, 0);
        __builtin_prefetch((const char*)(g_h0r + nb) + lane * 64, 0, 0);
      }
    }
    __syncthreads();

    // -------- A operands: contiguous-k 16-bit 16x32 layout (same k order as B) ---------
    // lane<16: M=lane,    elems 0..15 = K c*32+0..15
    // lane>=16: M=lane-16, elems 0..15 = K c*32+16..31
    v16h a[KC];
    #pragma unroll
    for (int c = 0; c < KC; ++c) {
      const _Float16* rp = &Atile[mrow * AW + c * 32 + hi16 * 16];
      const v8h lo = *(const v8h*)(rp);
      const v8h hv = *(const v8h*)(rp + 8);
      a[c] = __builtin_shufflevector(lo, hv, 0,1,2,3,4,5,6,7,8,9,10,11,12,13,14,15);
    }
    __syncthreads();

    const bool last = (t == Tloop - 1);
    #pragma unroll
    for (int k = 0; k < 3; ++k) {
      auto mm = [&](int tile) -> v8f {
        v8f acc;
        #pragma unroll
        for (int e = 0; e < 8; ++e) acc[e] = bias[tile];
        #pragma unroll
        for (int c = 0; c < KC; ++c) {
          const v16h b = *(const v16h*)&Wlds[(tile * KC + c) * 512 + lane * 16];
          acc = __builtin_amdgcn_wmma_f32_16x16x32_f16(
              false, a[c], false, b, (short)0, acc, false, false);
        }
        return acc;
      };
      const v8f zi = mm(k), zf = mm(3 + k), zg = mm(6 + k), zo = mm(9 + k);

      const int j = k * 16 + mrow;              // hidden unit (D-layout col)
      float hn[8];
      #pragma unroll
      for (int e = 0; e < 8; ++e) {
        const float iv = fast_sig(zi[e]);
        const float fv = fast_sig(zf[e]);
        const float gv = fast_tanh(zg[e]);
        const float ov = fast_sig(zo[e]);
        const float cv = fv * cst[k][e] + iv * gv;
        cst[k][e] = cv;
        hn[e] = ov * fast_tanh(cv);
      }
      #pragma unroll
      for (int e = 0; e < 8; ++e)               // new h -> A-tile cols < 48
        Atile[(e + hi16 * 8) * AW + j] = (_Float16)hn[e];
      if (LAYER == 1 && last && j < H36) {
        #pragma unroll
        for (int e = 0; e < 8; ++e)
          h1out[(size_t)(b0 + e + hi16 * 8) * H36 + j] = hn[e];
      }
    }

    if (LAYER == 0) {                           // stream h_t (f16) to global
      __syncthreads();
      const uint32_t* s = (const uint32_t*)Atile;
      const size_t base = ((size_t)tt * NBT + bt) * 384;
      for (int i = lane; i < 384; i += 32) {
        const int r = i / 24, d = i % 24;
        h0out[base + i] = s[r * 64 + d];
      }
    }
  }
}

// ------------------------------- FC head -----------------------------------
__global__ void fc_kernel(const float* __restrict__ h1f,
                          const float* __restrict__ h1r,
                          const float* __restrict__ fcW,
                          const float* __restrict__ fcb,
                          float* __restrict__ out)
{
  const int b = blockIdx.x * blockDim.x + threadIdx.x;
  if (b >= BTOT) return;
  float s = fcb[0];
  #pragma unroll 4
  for (int j = 0; j < H36; ++j) s += fcW[j]        * h1f[b * H36 + j];
  #pragma unroll 4
  for (int j = 0; j < H36; ++j) s += fcW[H36 + j]  * h1r[b * H36 + j];
  out[b] = s;
}

// ------------------------------- launch ------------------------------------
extern "C" void kernel_launch(void* const* d_in, const int* in_sizes, int n_in,
                              void* d_out, int out_size, void* d_ws, size_t ws_size,
                              hipStream_t stream)
{
  (void)in_sizes; (void)n_in; (void)out_size; (void)ws_size;
  const float* x     = (const float*)d_in[0];
  const float* Wih0f = (const float*)d_in[1];
  const float* Whh0f = (const float*)d_in[2];
  const float* bih0f = (const float*)d_in[3];
  const float* bhh0f = (const float*)d_in[4];
  const float* Wih0r = (const float*)d_in[5];
  const float* Whh0r = (const float*)d_in[6];
  const float* bih0r = (const float*)d_in[7];
  const float* bhh0r = (const float*)d_in[8];
  const float* Wih1f = (const float*)d_in[9];
  const float* Whh1f = (const float*)d_in[10];
  const float* bih1f = (const float*)d_in[11];
  const float* bhh1f = (const float*)d_in[12];
  const float* Wih1r = (const float*)d_in[13];
  const float* Whh1r = (const float*)d_in[14];
  const float* bih1r = (const float*)d_in[15];
  const float* bhh1r = (const float*)d_in[16];
  const float* fcW   = (const float*)d_in[17];
  const float* fcb   = (const float*)d_in[18];

  // workspace carve (~202 MB): packed weights | bias | h0f | h0r | h1f | h1r
  char* ws = (char*)d_ws;
  _Float16* wpack = (_Float16*)ws;
  float*    biasp = (float*)(ws + (size_t)4 * WSLOT * 2);
  size_t off = (size_t)4 * WSLOT * 2 + 4 * 192 * 4;          // 199680 (256-aligned)
  const size_t h0bytes = (size_t)TSEQ * NBT * 384 * 4;       // 100.7 MB each
  uint32_t* h0f = (uint32_t*)(ws + off); off += h0bytes;
  uint32_t* h0r = (uint32_t*)(ws + off); off += h0bytes;
  float*    h1f = (float*)(ws + off);    off += (size_t)BTOT * H36 * 4;
  float*    h1r = (float*)(ws + off);

  pack_weights<<<4, 256, 0, stream>>>(
      Wih0f, Whh0f, bih0f, bhh0f,  Wih0r, Whh0r, bih0r, bhh0r,
      Wih1f, Whh1f, bih1f, bhh1f,  Wih1r, Whh1r, bih1r, bhh1r,
      wpack, biasp);

  lstm_scan<0><<<NBT, 32, 0, stream>>>(x, nullptr, nullptr,
      wpack + (size_t)0 * WSLOT, biasp + 0,   h0f, nullptr, 0, TSEQ);
  lstm_scan<0><<<NBT, 32, 0, stream>>>(x, nullptr, nullptr,
      wpack + (size_t)1 * WSLOT, biasp + 192, h0r, nullptr, 1, TSEQ);
  lstm_scan<1><<<NBT, 32, 0, stream>>>(nullptr, h0f, h0r,
      wpack + (size_t)2 * WSLOT, biasp + 384, nullptr, h1f, 0, TSEQ);
  // out[:, -1] of the reverse direction == its first scan step -> 1 step only
  lstm_scan<1><<<NBT, 32, 0, stream>>>(nullptr, h0f, h0r,
      wpack + (size_t)3 * WSLOT, biasp + 576, nullptr, h1r, 1, 1);

  fc_kernel<<<(BTOT + 255) / 256, 256, 0, stream>>>(h1f, h1r, fcW, fcb,
                                                    (float*)d_out);
}
